// MultiHeadAttention_11252814315745
// MI455X (gfx1250) — matrix operand order repaired and checked by reading, not compiled, not run on hardware
//
#include <hip/hip_runtime.h>
#include <hip/hip_bf16.h>

// ---------------------------------------------------------------------------
// MI455X (gfx1250) multi-head attention forward.
// B=4, S=2048, E=1024, H=16, D=64.  fp32 in/out, bf16 WMMA internally.
//   0) one-shot fp32->bf16 conversion of x and all weights (bandwidth pass)
//   1) fused Q/K/V projection GEMM: all-bf16 operands, double-buffered LDS
//      staged entirely with global_load_async_to_lds_b128 (ASYNCcnt pipeline)
//   2) causal flash attention: WMMA QK^T / P@V, DPP softmax reductions,
//      async K-tile staging
//   3) output projection GEMM (same async pipeline) -> fp32
// ---------------------------------------------------------------------------

typedef __bf16 v8bf  __attribute__((ext_vector_type(8)));
typedef __bf16 v16bf __attribute__((ext_vector_type(16)));
typedef float  v8f   __attribute__((ext_vector_type(8)));

#define WMMA_BF16(A, B, C) \
    __builtin_amdgcn_wmma_f32_16x16x32_bf16(false, (A), false, (B), (short)0, (C), false, false)

// Build a 16-element bf16 fragment from two 16-byte chunks (ds_load_b128 x2).
__device__ inline v16bf ld_frag(const __bf16* lo, const __bf16* hi) {
    v8bf a = *(const v8bf*)lo;
    v8bf b = *(const v8bf*)hi;
    return __builtin_shufflevector(a, b, 0, 1, 2, 3, 4, 5, 6, 7, 8, 9, 10, 11, 12, 13, 14, 15);
}

// ---- DPP16 16-lane butterfly reductions (stay within each wave32 half) ----
template <int CTRL>
__device__ inline float dpp_xf(float x) {
    return __builtin_bit_cast(float,
        __builtin_amdgcn_update_dpp(0, __builtin_bit_cast(int, x), CTRL, 0xF, 0xF, true));
}
__device__ inline float red_max16(float x) {
    x = fmaxf(x, dpp_xf<0xB1>(x));   // quad_perm [1,0,3,2]   : xor 1
    x = fmaxf(x, dpp_xf<0x4E>(x));   // quad_perm [2,3,0,1]   : xor 2
    x = fmaxf(x, dpp_xf<0x141>(x));  // row_half_mirror       : combines 4-groups
    x = fmaxf(x, dpp_xf<0x140>(x));  // row_mirror            : combines 8-groups
    return x;
}
__device__ inline float red_sum16(float x) {
    x += dpp_xf<0xB1>(x);
    x += dpp_xf<0x4E>(x);
    x += dpp_xf<0x141>(x);
    x += dpp_xf<0x140>(x);
    return x;
}

// Async bf16 tile copy: global -> LDS, 16B per lane, tracked by ASYNCcnt.
__device__ inline void async_copy_b128(unsigned lds_off, const void* gaddr) {
    asm volatile("global_load_async_to_lds_b128 %0, %1, off"
                 :: "v"(lds_off), "v"((unsigned long long)(uintptr_t)gaddr)
                 : "memory");
}

// ---------------------------------------------------------------------------
// One-shot fp32 -> bf16 conversion (8 elements / thread).
// ---------------------------------------------------------------------------
__global__ __launch_bounds__(256) void cvt_f32_bf16_kernel(
    const float* __restrict__ in, __bf16* __restrict__ out) {
    size_t base = ((size_t)blockIdx.x * 256 + threadIdx.x) * 8;
    const float4 a = *(const float4*)(in + base);
    const float4 b = *(const float4*)(in + base + 4);
    v8bf o = {(__bf16)a.x, (__bf16)a.y, (__bf16)a.z, (__bf16)a.w,
              (__bf16)b.x, (__bf16)b.y, (__bf16)b.z, (__bf16)b.w};
    *(v8bf*)(out + base) = o;
}

// ---------------------------------------------------------------------------
// GEMM: C_q[M,N] = A[M,K] @ W_q[N,K]^T for q in [0,NMAT), all operands bf16.
// Block tile 128x64 per weight, 8 waves (4x2), wave tile 32x32 per weight.
// Double-buffered LDS, staged with async global->LDS; the ASYNCcnt wait is
// pipelined: after issuing the next tile's NLD loads, waiting down to NLD
// guarantees the previous tile is resident (async loads complete in order).
// ---------------------------------------------------------------------------
template <int NMAT, bool OUT_HEAD>
__global__ __launch_bounds__(256) void gemm_wmma_kernel(
    const __bf16* __restrict__ A,
    const __bf16* __restrict__ W0, const __bf16* __restrict__ W1,
    const __bf16* __restrict__ W2,
    void* __restrict__ C0, void* __restrict__ C1, void* __restrict__ C2,
    int M, int N, int K, int S) {
    constexpr int LDA = 40;           // padded LDS stride (80B, 16B-aligned)
    constexpr int NLD = 2 + NMAT;     // async instructions per thread per stage
    __shared__ __bf16 As[2][128 * LDA];
    __shared__ __bf16 Bs[2][NMAT][64 * LDA];

    const __bf16* Ws[3] = {W0, W1, W2};
    void*         Cs[3] = {C0, C1, C2};

    const int tid = threadIdx.x;
    const int lane = tid & 31, w = tid >> 5;
    const int lane16 = lane & 15, laneHi = lane >> 4;
    const int wm = w >> 1, wn = w & 1;  // 4 x 2 wave grid
    const int bm = blockIdx.y * 128;
    const int bn = blockIdx.x * 64;

    v8f acc[NMAT][2][2] = {};

    // Issue all async loads for one 32-wide k-slice into LDS buffer `buf`.
    auto stage = [&](int k0, int buf) {
        unsigned aB = (unsigned)(uintptr_t)(void*)&As[buf][0];
#pragma unroll
        for (int i = 0; i < 2; ++i) {           // A: 512 x 16B chunks
            int c = tid + i * 256;
            int r = c >> 2, col = (c & 3) * 8;
            async_copy_b128(aB + (unsigned)(r * LDA + col) * 2,
                            A + (size_t)(bm + r) * K + k0 + col);
        }
#pragma unroll
        for (int q = 0; q < NMAT; ++q) {        // each W: 256 x 16B chunks
            unsigned bB = (unsigned)(uintptr_t)(void*)&Bs[buf][q][0];
            int r = tid >> 2, col = (tid & 3) * 8;
            async_copy_b128(bB + (unsigned)(r * LDA + col) * 2,
                            Ws[q] + (size_t)(bn + r) * K + k0 + col);
        }
    };

    stage(0, 0);
    int buf = 0;
    for (int k0 = 0; k0 < K; k0 += 32, buf ^= 1) {
        if (k0 + 32 < K) {
            stage(k0 + 32, buf ^ 1);
            // prev batch (current buf) done once <= NLD newest remain in flight
            asm volatile("s_wait_asynccnt %0" :: "i"(NLD) : "memory");
        } else {
            asm volatile("s_wait_asynccnt 0x0" ::: "memory");
        }
        __syncthreads();

        // ---- A fragments loaded once, reused across NMAT weight matrices ----
        v16bf afr[2];
#pragma unroll
        for (int i = 0; i < 2; ++i) {
            int row = wm * 32 + i * 16 + lane16;
            const __bf16* p = &As[buf][row * LDA + (laneHi ? 8 : 0)];
            afr[i] = ld_frag(p, p + 16);        // K {0..7,16..23} / {8..15,24..31}
        }
#pragma unroll
        for (int q = 0; q < NMAT; ++q) {
            v16bf bfr[2];
#pragma unroll
            for (int j = 0; j < 2; ++j) {
                int n = wn * 32 + j * 16 + lane16;
                const __bf16* p = &Bs[buf][q][n * LDA + laneHi * 8];
                bfr[j] = ld_frag(p, p + 16);    // K {0..7,16..23} / {8..15,24..31}
            }
#pragma unroll
            for (int i = 0; i < 2; ++i)
#pragma unroll
                for (int j = 0; j < 2; ++j)
                    acc[q][i][j] = WMMA_BF16(afr[i], bfr[j], acc[q][i][j]);
        }
        __syncthreads();   // all reads of `buf` done before it is re-targeted
    }

    // ---- epilogue: C frag layout = (vgpr r, half) -> row, lane16 -> col ----
#pragma unroll
    for (int q = 0; q < NMAT; ++q) {
#pragma unroll
        for (int i = 0; i < 2; ++i) {
#pragma unroll
            for (int j = 0; j < 2; ++j) {
                int col = bn + wn * 32 + j * 16 + lane16;
#pragma unroll
                for (int r = 0; r < 8; ++r) {
                    int mI = bm + wm * 32 + i * 16 + r + laneHi * 8;
                    float val = acc[q][i][j][r];
                    if constexpr (OUT_HEAD) {
                        int bI = mI / S, sI = mI - bI * S;
                        int hI = col >> 6, dI = col & 63;
                        ((__bf16*)Cs[q])[(((size_t)bI * 16 + hI) * S + sI) * 64 + dI] = (__bf16)val;
                    } else {
                        ((float*)Cs[q])[(size_t)mI * N + col] = val;
                    }
                }
            }
        }
    }
}

// ---------------------------------------------------------------------------
// Causal flash attention.  Q/K/V bf16 in (B,H,S,64).  Output bf16 (B,S,E).
// 128 threads = 4 waves; wave w handles 16 query rows; WG covers 64 rows.
// ---------------------------------------------------------------------------
__global__ __launch_bounds__(128) void flash_attn_kernel(
    const __bf16* __restrict__ Q, const __bf16* __restrict__ Kg,
    const __bf16* __restrict__ Vg, __bf16* __restrict__ Obuf, int S) {
    constexpr int LDK = 72;  // Ks stride (144B, 16B-aligned)
    constexpr int LDP = 40;  // Vt / Ps stride (80B, 16B-aligned)
    __shared__ __bf16 Ks[32 * LDK];      // K block, row-major [key][d]
    __shared__ __bf16 Vt[64 * LDP];      // V block, transposed [d][key]
    __shared__ __bf16 Ps[4][16 * LDP];   // per-wave P tile 16x32

    const int tid = threadIdx.x;
    const int lane = tid & 31, w = tid >> 5;
    const int lane16 = lane & 15, laneHi = lane >> 4;
    const int bh = blockIdx.y;                 // b*16 + h
    const int b = bh >> 4, h = bh & 15;
    const int qbase = blockIdx.x * 64 + w * 16;

    const __bf16* Qp = Q + (size_t)bh * S * 64;
    const __bf16* Kp = Kg + (size_t)bh * S * 64;
    const __bf16* Vp = Vg + (size_t)bh * S * 64;

    const unsigned ksBase = (unsigned)(uintptr_t)(void*)Ks;  // LDS byte offset

    // Preload Q fragments for both 32-wide d-blocks (D=64 -> 2 WMMA k-steps).
    v16bf qf[2];
    {
        int qrow = qbase + lane16;
#pragma unroll
        for (int db = 0; db < 2; ++db) {
            const __bf16* p = Qp + (size_t)qrow * 64 + db * 32 + (laneHi ? 8 : 0);
            qf[db] = ld_frag(p, p + 16);
        }
    }

    float m_run[8], l_run[8];
#pragma unroll
    for (int r = 0; r < 8; ++r) { m_run[r] = -1e30f; l_run[r] = 0.f; }
    v8f o_acc[4] = {};

    const float sscale = 0.125f * 1.44269504088896f;  // 1/sqrt(64) * log2(e)
    const int nkb = (blockIdx.x * 64 + 64) >> 5;      // causal key-block bound

    for (int kb = 0; kb < nkb; ++kb) {
        const int kstart = kb * 32;
        __syncthreads();
        // ---- stage K block (32x64) via async global->LDS (bf16 passthrough) ----
#pragma unroll
        for (int i = 0; i < 2; ++i) {
            int c = tid + i * 128;              // 256 chunks of 8 bf16 (16B)
            int r = c >> 3, col = (c & 7) * 8;
            async_copy_b128(ksBase + (unsigned)(r * LDK + col) * 2,
                            Kp + (size_t)(kstart + r) * 64 + col);
        }
        // ---- stage V block transposed: Vt[d][key] (register path) ----
        {
            int n = tid >> 2, d0 = (tid & 3) * 16;
            v8bf a = *(const v8bf*)(Vp + (size_t)(kstart + n) * 64 + d0);
            v8bf c = *(const v8bf*)(Vp + (size_t)(kstart + n) * 64 + d0 + 8);
#pragma unroll
            for (int e = 0; e < 8; ++e) Vt[(d0 + e) * LDP + n] = a[e];
#pragma unroll
            for (int e = 0; e < 8; ++e) Vt[(d0 + 8 + e) * LDP + n] = c[e];
        }
        asm volatile("s_wait_asynccnt 0x0" ::: "memory");
        __syncthreads();

        // ---- scores: S16x32 = Q(16x64) @ K^T, 2 n-subtiles x 2 k-steps ----
        v8f sc[2] = {};
#pragma unroll
        for (int nt = 0; nt < 2; ++nt) {
#pragma unroll
            for (int db = 0; db < 2; ++db) {
                int n = nt * 16 + lane16;       // key index within block
                const __bf16* p = &Ks[n * LDK + db * 32 + laneHi * 8];
                v16bf kf = ld_frag(p, p + 16);
                sc[nt] = WMMA_BF16(qf[db], kf, sc[nt]);
            }
        }

        // ---- online softmax in C-fragment geometry (DPP reductions) ----
#pragma unroll
        for (int r = 0; r < 8; ++r) {
            int grow = qbase + r + laneHi * 8;
#pragma unroll
            for (int nt = 0; nt < 2; ++nt) {
                int col = kstart + nt * 16 + lane16;
                float xv = sc[nt][r] * sscale;
                if (col > grow) xv = -3.0e38f;  // causal mask
                sc[nt][r] = xv;
            }
            float mx = red_max16(fmaxf(sc[0][r], sc[1][r]));
            float mnew = fmaxf(m_run[r], mx);
            float alpha = exp2f(m_run[r] - mnew);
            float rs = 0.f;
#pragma unroll
            for (int nt = 0; nt < 2; ++nt) {
                float pv = exp2f(sc[nt][r] - mnew);
                rs += pv;
                Ps[w][(r + laneHi * 8) * LDP + nt * 16 + lane16] = (__bf16)pv;
            }
            rs = red_sum16(rs);
            l_run[r] = l_run[r] * alpha + rs;
            m_run[r] = mnew;
#pragma unroll
            for (int j = 0; j < 4; ++j) o_acc[j][r] = o_acc[j][r] * alpha;
        }
        // wave-local LDS store->load ordering (DS ops are in-order per wave)
        asm volatile("s_wait_dscnt 0x0" ::: "memory");

        // ---- O += P(16x32) @ V(32x64): A-frag from Ps, B-frags from Vt ----
        {
            const __bf16* pp = &Ps[w][lane16 * LDP + (laneHi ? 8 : 0)];
            v16bf pf = ld_frag(pp, pp + 16);
#pragma unroll
            for (int j = 0; j < 4; ++j) {
                const __bf16* vp = &Vt[(j * 16 + lane16) * LDP + laneHi * 8];
                v16bf vf = ld_frag(vp, vp + 16);
                o_acc[j] = WMMA_BF16(pf, vf, o_acc[j]);
            }
        }
    }

    // ---- normalize and write (B,S,E) bf16 ----
#pragma unroll
    for (int j = 0; j < 4; ++j) {
        int col = h * 64 + j * 16 + lane16;
#pragma unroll
        for (int r = 0; r < 8; ++r) {
            int srow = qbase + r + laneHi * 8;
            float val = o_acc[j][r] / l_run[r];
            Obuf[((size_t)b * S + srow) * 1024 + col] = (__bf16)val;
        }
    }
}

// ---------------------------------------------------------------------------
extern "C" void kernel_launch(void* const* d_in, const int* in_sizes, int n_in,
                              void* d_out, int out_size, void* d_ws, size_t ws_size,
                              hipStream_t stream) {
    const float* x  = (const float*)d_in[0];
    const float* Wq = (const float*)d_in[1];
    const float* Wk = (const float*)d_in[2];
    const float* Wv = (const float*)d_in[3];
    const float* Wo = (const float*)d_in[4];

    const int B = 4, S = 2048, E = 1024;
    const int M = B * S;                       // 8192
    const size_t per = (size_t)M * E;          // activation-sized bf16 buffer
    const size_t wper = (size_t)E * E;         // weight-sized bf16 buffer

    __bf16* Xb  = (__bf16*)d_ws;               // x  bf16 (B*S, E)
    __bf16* Wqb = Xb + per;                    // weights bf16 (E, E)
    __bf16* Wkb = Wqb + wper;
    __bf16* Wvb = Wkb + wper;
    __bf16* Wob = Wvb + wper;
    __bf16* Qb  = Wob + wper;                  // (B,H,S,D)
    __bf16* Kb  = Qb + per;
    __bf16* Vb  = Kb + per;
    __bf16* Ob  = Vb + per;                    // (B,S,E)

    // 0) one-shot fp32 -> bf16 conversions (8 elems / thread)
    cvt_f32_bf16_kernel<<<per / (256 * 8), 256, 0, stream>>>(x, Xb);
    cvt_f32_bf16_kernel<<<wper / (256 * 8), 256, 0, stream>>>(Wq, Wqb);
    cvt_f32_bf16_kernel<<<wper / (256 * 8), 256, 0, stream>>>(Wk, Wkb);
    cvt_f32_bf16_kernel<<<wper / (256 * 8), 256, 0, stream>>>(Wv, Wvb);
    cvt_f32_bf16_kernel<<<wper / (256 * 8), 256, 0, stream>>>(Wo, Wob);

    dim3 gg(E / 64, M / 128);                  // 16 x 64 blocks
    // 1) fused Q/K/V projection: x staged once per tile, 3 weight matrices.
    gemm_wmma_kernel<3, true><<<gg, 256, 0, stream>>>(
        Xb, Wqb, Wkb, Wvb, Qb, Kb, Vb, M, E, E, S);

    // 2) causal flash attention
    dim3 ga(S / 64, B * 16);                   // 32 x 64 blocks, 128 threads
    flash_attn_kernel<<<ga, 128, 0, stream>>>(Qb, Kb, Vb, Ob, S);

    // 3) output projection -> fp32 d_out.
    gemm_wmma_kernel<1, false><<<gg, 256, 0, stream>>>(
        Ob, Wob, nullptr, nullptr, d_out, nullptr, nullptr, M, E, E, S);
}
